// MultiHeadAttentionMap_19662360281783
// MI455X (gfx1250) — matrix operand order repaired and checked by reading, not compiled, not run on hardware
//
#include <hip/hip_runtime.h>

typedef __attribute__((ext_vector_type(16))) __bf16 v16bf;
typedef __attribute__((ext_vector_type(8)))  __bf16 v8bf;
typedef __attribute__((ext_vector_type(8)))  float  v8f;
typedef __attribute__((ext_vector_type(4)))  unsigned int v4u;
typedef __attribute__((ext_vector_type(8)))  int v8i;
typedef __attribute__((ext_vector_type(4)))  int v4i;

#define S_LEN    2048
#define D_MODEL  1024
#define NHEADS   16
#define HEAD_DIM 64
// softmax((logits)*0.125): fold scale and log2(e) into one exp2 factor
#define SCL2E    0.18033688011112042592f  /* 0.125 * log2(e) */

// ---- 16-lane reductions in pure VALU via DPP16 (no LDS round trips) --------
// xor1 -> xor2 -> half-row mirror -> row mirror: after each step the pairing
// partner group is already uniform, so mirror works as well as xor.
template <int CTRL>
__device__ __forceinline__ float dpp_mov(float x) {
  int t = __builtin_amdgcn_update_dpp(0, __builtin_bit_cast(int, x), CTRL, 0xF, 0xF, true);
  return __builtin_bit_cast(float, t);
}
__device__ __forceinline__ float red_max16(float x) {
  x = fmaxf(x, dpp_mov<0x0B1>(x));  // quad_perm [1,0,3,2]
  x = fmaxf(x, dpp_mov<0x04E>(x));  // quad_perm [2,3,0,1]
  x = fmaxf(x, dpp_mov<0x141>(x));  // row_half_mirror
  x = fmaxf(x, dpp_mov<0x140>(x));  // row_mirror
  return x;
}
__device__ __forceinline__ float red_sum16(float x) {
  x += dpp_mov<0x0B1>(x);
  x += dpp_mov<0x04E>(x);
  x += dpp_mov<0x141>(x);
  x += dpp_mov<0x140>(x);
  return x;
}

__device__ __forceinline__ v16bf cat16(v8bf lo, v8bf hi) {
  return __builtin_shufflevector(lo, hi, 0, 1, 2, 3, 4, 5, 6, 7,
                                 8, 9, 10, 11, 12, 13, 14, 15);
}
__device__ __forceinline__ v8bf cvt8(float4 a, float4 b) {
  v8bf r;
  r[0] = (__bf16)a.x; r[1] = (__bf16)a.y; r[2] = (__bf16)a.z; r[3] = (__bf16)a.w;
  r[4] = (__bf16)b.x; r[5] = (__bf16)b.y; r[6] = (__bf16)b.z; r[7] = (__bf16)b.w;
  return r;
}

// Issue a 2D TDM tile load (f32 elements) into LDS. One call per wave.
// 6-arg builtin form: (v4u, v8i, v4i, v4i, v8i, i32 cpol).
__device__ __forceinline__ void tdm_load_2d_f32(unsigned int ldsAddr, const void* gptr,
                                                unsigned int tensor_w, unsigned int tensor_h,
                                                unsigned int tile_w, unsigned int tile_h,
                                                unsigned int stride_elems) {
  unsigned long long ga = (unsigned long long)(uintptr_t)gptr;
  v4u g0;
  g0[0] = 1u;                                   // count=1 valid user D#
  g0[1] = ldsAddr;                              // LDS byte address
  g0[2] = (unsigned int)ga;                     // global_addr[31:0]
  g0[3] = (unsigned int)((ga >> 32) & 0x1ffffffu) | (2u << 30);  // addr[56:32] | type=2
  v8i g1;
  g1[0] = (int)(2u << 16);                      // data_size=2 (4 bytes); no multicast
  g1[1] = (int)((tensor_w & 0xffffu) << 16);    // tensor_dim0[15:0] @ bit48
  g1[2] = (int)((tensor_w >> 16) | ((tensor_h & 0xffffu) << 16));
  g1[3] = (int)((tensor_h >> 16) | (tile_w << 16));   // tile_dim0 @ bit112
  g1[4] = (int)tile_h;                          // tile_dim1 (tile_dim2 = 0)
  g1[5] = (int)stride_elems;                    // tensor_dim0_stride[31:0]
  g1[6] = 0;
  g1[7] = 0;
  v4i gz4;  gz4[0] = 0; gz4[1] = 0; gz4[2] = 0; gz4[3] = 0;
  v8i gz8;
#pragma unroll
  for (int i = 0; i < 8; ++i) gz8[i] = 0;
  __builtin_amdgcn_tensor_load_to_lds(g0, g1, gz4, gz4, gz8, 0);
}

// ---------------------------------------------------------------------------
// Kernel 1: flash attention with bias. Block = 8 waves = 128 query rows of one
// head. Bias tiles stream in via TDM; K/V tiles staged cooperatively in LDS as
// bf16. Each wave owns a 16-row q tile, walks keys in 32-wide blocks.
// ---------------------------------------------------------------------------
__global__ __launch_bounds__(256) void flash_attn_bias_kernel(
    const float* __restrict__ Q, const float* __restrict__ K,
    const float* __restrict__ V, const float* __restrict__ bias,
    __bf16* __restrict__ combined) {
  __shared__ float  ldsBias[128 * 32];   // 16 KB, TDM destination
  __shared__ __bf16 ldsKT[64 * 32];      // 4 KB, K tile transposed [hd][key]
  __shared__ __bf16 ldsV[32 * 64];       // 4 KB, V tile natural   [key][hd]
  __shared__ __bf16 pbuf[8][16 * 32];    // 8 KB, per-wave P relayout C->A

  const int tid  = threadIdx.x;
  const int wave = tid >> 5;
  const int lane = tid & 31;
  const int mrow = lane & 15;
  const int lh   = lane >> 4;            // lane half

  const int h  = blockIdx.y;
  const int b  = blockIdx.z;
  const int q0 = blockIdx.x * 128 + wave * 16;

  const float* Qb = Q + ((size_t)b * S_LEN) * D_MODEL + (size_t)h * HEAD_DIM;
  const float* Kb = K + ((size_t)b * S_LEN) * D_MODEL + (size_t)h * HEAD_DIM;
  const float* Vb = V + ((size_t)b * S_LEN) * D_MODEL + (size_t)h * HEAD_DIM;
  const float* Bblk = bias + ((size_t)h * S_LEN + (size_t)(blockIdx.x * 128)) * S_LEN;

  const unsigned int biasLds = (unsigned int)(uintptr_t)&ldsBias[0];

  // staging assignment: 8 threads cover one key row (64 floats)
  const int skey   = tid >> 3;       // 0..31
  const int sdbase = (tid & 7) * 8;  // 0..56

  // ---- Q tile (16 x 64) as two 16x32 A operands, vectorized loads ----
  v16bf aQ[2];
#pragma unroll
  for (int c = 0; c < 2; ++c) {
    const float* qr = &Qb[(size_t)(q0 + mrow) * D_MODEL + 32 * c + 8 * lh];
    float4 a0 = ((const float4*)qr)[0];
    float4 a1 = ((const float4*)qr)[1];
    float4 a2 = ((const float4*)(qr + 16))[0];
    float4 a3 = ((const float4*)(qr + 16))[1];
    aQ[c] = cat16(cvt8(a0, a1), cvt8(a2, a3));
  }

  float mrun[8], lrun[8];   // running row max (raw-logit units) / row sum
  v8f acc[4];
#pragma unroll
  for (int r = 0; r < 8; ++r) { mrun[r] = -1e30f; lrun[r] = 0.0f; }
#pragma unroll
  for (int nc = 0; nc < 4; ++nc)
#pragma unroll
    for (int r = 0; r < 8; ++r) acc[nc][r] = 0.0f;

  for (int kb = 0; kb < S_LEN; kb += 32) {
    __syncthreads();  // previous iteration's consumers are done with LDS tiles

    // ---- TDM: bias tile [128 rows x 32 cols] f32 -> LDS (one DMA per block)
    if (wave == 0)
      tdm_load_2d_f32(biasLds, Bblk + kb, S_LEN, S_LEN, 32, 128, S_LEN);

    // ---- cooperative K/V tile staging (32 keys x 64 hd), f32->bf16 ----
    {
      const float* kr = &Kb[(size_t)(kb + skey) * D_MODEL + sdbase];
      float4 k0 = ((const float4*)kr)[0];
      float4 k1 = ((const float4*)kr)[1];
      const float* vr = &Vb[(size_t)(kb + skey) * D_MODEL + sdbase];
      float4 v0 = ((const float4*)vr)[0];
      float4 v1 = ((const float4*)vr)[1];
      // K transposed: [hd][key]
      ldsKT[(sdbase + 0) * 32 + skey] = (__bf16)k0.x;
      ldsKT[(sdbase + 1) * 32 + skey] = (__bf16)k0.y;
      ldsKT[(sdbase + 2) * 32 + skey] = (__bf16)k0.z;
      ldsKT[(sdbase + 3) * 32 + skey] = (__bf16)k0.w;
      ldsKT[(sdbase + 4) * 32 + skey] = (__bf16)k1.x;
      ldsKT[(sdbase + 5) * 32 + skey] = (__bf16)k1.y;
      ldsKT[(sdbase + 6) * 32 + skey] = (__bf16)k1.z;
      ldsKT[(sdbase + 7) * 32 + skey] = (__bf16)k1.w;
      // V natural: [key][hd], one 16B vector store
      *(v8bf*)&ldsV[skey * 64 + sdbase] = cvt8(v0, v1);
    }

    if (wave == 0) __builtin_amdgcn_s_wait_tensorcnt(0);
    __syncthreads();  // bias + K + V tiles visible to all waves

    // ---- logits for two 16-key subtiles: z = Q*K^T + bias (bias rides in C)
    v8f z[2];
#pragma unroll
    for (int sub = 0; sub < 2; ++sub) {
      v8f cb;
#pragma unroll
      for (int r = 0; r < 8; ++r)
        cb[r] = ldsBias[(wave * 16 + r + 8 * lh) * 32 + 16 * sub + mrow];
      v16bf bK[2];
#pragma unroll
      for (int c = 0; c < 2; ++c) {
#pragma unroll
        for (int s = 0; s < 8; ++s) {
          bK[c][s]     = ldsKT[(32 * c + 8 * lh + s) * 32 + 16 * sub + mrow];
          bK[c][8 + s] = ldsKT[(32 * c + 16 + 8 * lh + s) * 32 + 16 * sub + mrow];
        }
      }
      v8f t = __builtin_amdgcn_wmma_f32_16x16x32_bf16(false, aQ[0], false, bK[0],
                                                      (short)0, cb, false, false);
      t = __builtin_amdgcn_wmma_f32_16x16x32_bf16(false, aQ[1], false, bK[1],
                                                  (short)0, t, false, false);
      z[sub] = t;
    }

    // ---- online softmax update (raw-logit units; scale folded into exp2) ----
    float pv0[8], pv1[8];
#pragma unroll
    for (int r = 0; r < 8; ++r) {
      float rm   = red_max16(fmaxf(z[0][r], z[1][r]));
      float mnew = fmaxf(mrun[r], rm);
      float alpha = __builtin_amdgcn_exp2f((mrun[r] - mnew) * SCL2E);
      float p0 = __builtin_amdgcn_exp2f((z[0][r] - mnew) * SCL2E);
      float p1 = __builtin_amdgcn_exp2f((z[1][r] - mnew) * SCL2E);
      float rs = red_sum16(p0 + p1);
      lrun[r] = lrun[r] * alpha + rs;
      mrun[r] = mnew;
      pv0[r] = p0; pv1[r] = p1;
#pragma unroll
      for (int nc = 0; nc < 4; ++nc) acc[nc][r] *= alpha;
    }

    // ---- stage P (16x32, C layout) into per-wave LDS, reload in A layout ----
#pragma unroll
    for (int r = 0; r < 8; ++r) {
      pbuf[wave][(r + 8 * lh) * 32 + mrow]      = (__bf16)pv0[r];
      pbuf[wave][(r + 8 * lh) * 32 + 16 + mrow] = (__bf16)pv1[r];
    }
    v16bf aP;
    {
      v8bf lo = *(const v8bf*)&pbuf[wave][mrow * 32 + 8 * lh];
      v8bf hi = *(const v8bf*)&pbuf[wave][mrow * 32 + 16 + 8 * lh];
      aP = cat16(lo, hi);
    }

    // ---- accumulate P @ V over 4 value chunks ----
#pragma unroll
    for (int nc = 0; nc < 4; ++nc) {
      v16bf bV;
#pragma unroll
      for (int s = 0; s < 8; ++s) {
        bV[s]     = ldsV[(8 * lh + s) * 64 + nc * 16 + mrow];
        bV[8 + s] = ldsV[(16 + 8 * lh + s) * 64 + nc * 16 + mrow];
      }
      acc[nc] = __builtin_amdgcn_wmma_f32_16x16x32_bf16(false, aP, false, bV,
                                                        (short)0, acc[nc], false, false);
    }
  }

  // ---- normalize, write combined (bf16) ----
#pragma unroll
  for (int r = 0; r < 8; ++r) {
    const float rl = 1.0f / lrun[r];
#pragma unroll
    for (int nc = 0; nc < 4; ++nc)
      combined[((size_t)b * S_LEN + q0 + r + 8 * lh) * D_MODEL +
               h * HEAD_DIM + nc * 16 + mrow] = (__bf16)(acc[nc][r] * rl);
  }
}

// ---------------------------------------------------------------------------
// Kernel 2: out = combined[4096,1024] @ w_out[1024,1024], f32 result.
// One wave computes a 16x64 output tile; block = 8 waves = 128 rows x 64 cols.
// ---------------------------------------------------------------------------
__global__ __launch_bounds__(256) void out_proj_kernel(
    const __bf16* __restrict__ Cmb, const float* __restrict__ Wout,
    float* __restrict__ Out) {
  const int wave = threadIdx.x >> 5;
  const int lane = threadIdx.x & 31;
  const int mrow = lane & 15;
  const int lh   = lane >> 4;
  const int m0 = blockIdx.x * 128 + wave * 16;
  const int n0 = blockIdx.y * 64;

  v8f acc[4];
#pragma unroll
  for (int nc = 0; nc < 4; ++nc)
#pragma unroll
    for (int r = 0; r < 8; ++r) acc[nc][r] = 0.0f;

  for (int kc = 0; kc < D_MODEL / 32; ++kc) {
    const __bf16* ar = &Cmb[(size_t)(m0 + mrow) * D_MODEL + kc * 32 + 8 * lh];
    v16bf aC = cat16(*(const v8bf*)ar, *(const v8bf*)(ar + 16));
#pragma unroll
    for (int nc = 0; nc < 4; ++nc) {
      const float* wr = &Wout[(size_t)(kc * 32) * D_MODEL + n0 + nc * 16 + mrow];
      v16bf bW;
#pragma unroll
      for (int s = 0; s < 8; ++s) {
        bW[s]     = (__bf16)wr[(size_t)(8 * lh + s) * D_MODEL];
        bW[8 + s] = (__bf16)wr[(size_t)(16 + 8 * lh + s) * D_MODEL];
      }
      acc[nc] = __builtin_amdgcn_wmma_f32_16x16x32_bf16(false, aC, false, bW,
                                                        (short)0, acc[nc], false, false);
    }
  }

#pragma unroll
  for (int r = 0; r < 8; ++r)
#pragma unroll
    for (int nc = 0; nc < 4; ++nc)
      Out[(size_t)(m0 + r + 8 * lh) * D_MODEL + n0 + nc * 16 + mrow] = acc[nc][r];
}

// ---------------------------------------------------------------------------
extern "C" void kernel_launch(void* const* d_in, const int* in_sizes, int n_in,
                              void* d_out, int out_size, void* d_ws, size_t ws_size,
                              hipStream_t stream) {
  const float* Q    = (const float*)d_in[0];
  const float* K    = (const float*)d_in[1];
  const float* V    = (const float*)d_in[2];
  const float* bias = (const float*)d_in[3];
  const float* Wout = (const float*)d_in[4];
  float* Out        = (float*)d_out;
  __bf16* combined  = (__bf16*)d_ws;   // B*S*D bf16 = 8.4 MB scratch

  const int B = in_sizes[0] / (S_LEN * D_MODEL);   // = 2

  dim3 g1(S_LEN / 128, NHEADS, B);
  flash_attn_bias_kernel<<<g1, 256, 0, stream>>>(Q, K, V, bias, combined);

  dim3 g2((B * S_LEN) / 128, D_MODEL / 64);
  out_proj_kernel<<<g2, 256, 0, stream>>>(combined, Wout, Out);
}
